// MixtureOfDepthsBlock_17927193493873
// MI455X (gfx1250) — matrix operand order repaired and checked by reading, not compiled, not run on hardware
//
#include <hip/hip_runtime.h>

// ---------------- problem constants ----------------
constexpr int Bc  = 4;
constexpr int S   = 4096;
constexpr int H   = 2048;
constexpr int DFF = 8192;
constexpr int BS  = Bc * S;          // 16384 tokens
constexpr int TOPK = S / 2;          // capacity 0.5 -> k = 2048
#define EPSV 1e-5f

typedef unsigned short u16;
typedef unsigned int   u32;

typedef __attribute__((ext_vector_type(16))) __bf16 v16bf;
typedef __attribute__((ext_vector_type(8)))  float  v8f;

struct alignas(16) Q4 { u32 x, y, z, w; };
struct alignas(16) F4 { float x, y, z, w; };

// ---------------- CDNA5 feature detection ----------------
#if defined(__has_builtin)
#if __has_builtin(__builtin_amdgcn_global_load_async_to_lds_b128)
#define HAVE_ASYNC 1
#endif
#if __has_builtin(__builtin_amdgcn_s_wait_asynccnt)
#define HAVE_WAIT_ASYNC 1
#endif
#if __has_builtin(__builtin_amdgcn_tensor_load_to_lds) && \
    __has_builtin(__builtin_amdgcn_s_wait_tensorcnt)
#define HAVE_TDM 1
#endif
#endif
#ifndef HAVE_ASYNC
#define HAVE_ASYNC 0
#endif
#ifndef HAVE_WAIT_ASYNC
#define HAVE_WAIT_ASYNC 0
#endif
#ifndef HAVE_TDM
#define HAVE_TDM 0
#endif

typedef __attribute__((address_space(3))) void LAS;

#if HAVE_ASYNC
// builtin prototype (from compiler diagnostic): takes v4i addrspace(1)* / addrspace(3)*
typedef int v4i_g __attribute__((vector_size(16)));
typedef __attribute__((address_space(1))) v4i_g GV4;
typedef __attribute__((address_space(3))) v4i_g LV4;
__device__ __forceinline__ void async_cp16(const void* g, void* l) {
  __builtin_amdgcn_global_load_async_to_lds_b128((GV4*)g, (LV4*)l, 0, 0);
}
#endif

__device__ __forceinline__ void wait_async0() {
#if HAVE_WAIT_ASYNC
  __builtin_amdgcn_s_wait_asynccnt(0);
#elif HAVE_ASYNC
  asm volatile("s_wait_asynccnt 0" ::: "memory");
#endif
}

#if HAVE_TDM
typedef __attribute__((ext_vector_type(4))) unsigned v4u;
typedef __attribute__((ext_vector_type(8))) int      v8i;
typedef __attribute__((ext_vector_type(4))) int      v4i;

// 2D TDM load: 128 rows x 64 bf16 elems from BT (row stride Kst elems) into
// LDS tile with stride-72 rows via descriptor padding (32 dwords -> +4 dwords).
__device__ __forceinline__ void tdm_load_bt(u16* sB, const u16* BT, int n0, int kb,
                                            int Kst, int Nd) {
  u32 lds = (u32)(size_t)(LAS*)sB;
  unsigned long long ga = (unsigned long long)(size_t)&BT[(size_t)n0 * Kst + kb];
  v4u g0 = { 1u,                                   // count=1, no gather
             lds,                                  // lds_addr (bytes)
             (u32)ga,
             ((u32)((ga >> 32) & 0x01FFFFFFull)) | 0x80000000u };  // type=2
  v8i g1 = { (int)((1u << 16)    // data_size = 2 bytes
                 | (1u << 20)    // pad_enable
                 | (4u << 22)    // pad_interval: 2^(4+1)=32 dwords (one row)
                 | (3u << 25)),  // pad_amount: 4 dwords = 8 elems -> stride 72
             (int)(((u32)Kst & 0xFFFFu) << 16),                       // tensor_dim0 lo
             (int)((((u32)Kst >> 16) & 0xFFFFu) | (((u32)Nd & 0xFFFFu) << 16)),
             (int)((((u32)Nd >> 16) & 0xFFFFu) | (64u << 16)),        // tile_dim0=64
             (int)128,                                                // tile_dim1=128
             (int)Kst,                                                // dim0_stride lo32
             0, 0 };
  v4i z4 = {0, 0, 0, 0};
#if __clang_major__ >= 23
  v8i z8 = {0, 0, 0, 0, 0, 0, 0, 0};
  __builtin_amdgcn_tensor_load_to_lds(g0, g1, z4, z4, z8, 0);
#else
  __builtin_amdgcn_tensor_load_to_lds(g0, g1, z4, z4, 0);
#endif
}
#endif

// fp32 -> bf16 round-to-nearest-even
__device__ __forceinline__ u16 f2bf(float f) {
  u32 u = __float_as_uint(f);
  u32 r = u + 0x7FFFu + ((u >> 16) & 1u);
  return (u16)(r >> 16);
}

__device__ __forceinline__ v16bf load_frag(const u16* p0, const u16* p1) {
  union { Q4 q[2]; v16bf v; } u;
  u.q[0] = *(const Q4*)p0;
  u.q[1] = *(const Q4*)p1;
  return u.v;
}

__device__ __forceinline__ float block_reduce_sum(float v) {
  __shared__ float red[256];
  red[threadIdx.x] = v;
  __syncthreads();
  for (int s = 128; s > 0; s >>= 1) {
    if ((int)threadIdx.x < s) red[threadIdx.x] += red[threadIdx.x + s];
    __syncthreads();
  }
  float r = red[0];
  __syncthreads();
  return r;
}

// ---------------- router: weights = sigmoid(x . w_r + b) ----------------
__global__ __launch_bounds__(256)
void k_router(const float* __restrict__ x, const float* __restrict__ wr,
              const float* __restrict__ br, float* __restrict__ weights,
              int* __restrict__ cnt) {
  const int t = blockIdx.x;
  float partial = 0.f;
  const float* row = x + (size_t)t * H;
  for (int i = threadIdx.x; i < H; i += 256) partial += row[i] * wr[i];
  float dot = block_reduce_sum(partial);
  if (threadIdx.x == 0) {
    float l = dot + br[0];
    weights[t] = 1.f / (1.f + __expf(-l));
    if (t == 0) *cnt = 0;
  }
}

// ---------------- exact k-th largest per batch (radix bit-select) ----------------
__global__ __launch_bounds__(256)
void k_select(const float* __restrict__ weights, float* __restrict__ thresh) {
  const int b = blockIdx.x;
  __shared__ u32 sw[S];
  __shared__ int scount;
  __shared__ u32 sprefix;
  for (int i = threadIdx.x; i < S; i += 256)
    sw[i] = __float_as_uint(weights[(size_t)b * S + i]);   // sigmoid>0 -> monotone bits
  u32 prefix = 0;
  for (int bit = 31; bit >= 0; --bit) {
    __syncthreads();
    if (threadIdx.x == 0) scount = 0;
    __syncthreads();
    u32 cand = prefix | (1u << bit);
    int local = 0;
    for (int i = threadIdx.x; i < S; i += 256) local += (sw[i] >= cand) ? 1 : 0;
    atomicAdd(&scount, local);
    __syncthreads();
    if (threadIdx.x == 0) sprefix = (scount >= TOPK) ? cand : prefix;
    __syncthreads();
    prefix = sprefix;
  }
  if (threadIdx.x == 0) thresh[b] = __uint_as_float(prefix);
}

// ---------------- compact selected tokens + RMSNorm1 -> bf16 ----------------
__global__ __launch_bounds__(256)
void k_compact_norm(const float* __restrict__ x, const float* __restrict__ weights,
                    const float* __restrict__ thresh, const float* __restrict__ n1w,
                    int* __restrict__ cnt, int* __restrict__ sel, u16* __restrict__ n1c) {
  const int t = blockIdx.x;
  const int b = t >> 12;                      // t / S
  float w  = weights[t];
  float th = thresh[b];
  if (w < th) return;                         // uniform per block
  __shared__ int spos;
  if (threadIdx.x == 0) spos = atomicAdd(cnt, 1);
  const float* row = x + (size_t)t * H;
  float partial = 0.f;
  for (int i = threadIdx.x; i < H; i += 256) { float v = row[i]; partial += v * v; }
  float ss = block_reduce_sum(partial);       // syncthreads inside makes spos visible
  float r = rsqrtf(ss / (float)H + EPSV);
  int pos = spos;
  if (threadIdx.x == 0) sel[pos] = t;
  u16* dst = n1c + (size_t)pos * H;
  for (int i = threadIdx.x; i < H; i += 256) dst[i] = f2bf(row[i] * r * n1w[i]);
}

// ---------------- RMSNorm2 over compacted residual ----------------
__global__ __launch_bounds__(256)
void k_norm2(const float* __restrict__ resc, const float* __restrict__ n2w,
             const int* __restrict__ cnt, u16* __restrict__ n2c) {
  const int m = blockIdx.x;
  if (m >= *cnt) return;
  const float* row = resc + (size_t)m * H;
  float partial = 0.f;
  for (int i = threadIdx.x; i < H; i += 256) { float v = row[i]; partial += v * v; }
  float ss = block_reduce_sum(partial);
  float r = rsqrtf(ss / (float)H + EPSV);
  u16* dst = n2c + (size_t)m * H;
  for (int i = threadIdx.x; i < H; i += 256) dst[i] = f2bf(row[i] * r * n2w[i]);
}

// ---------------- out = x (background for unselected tokens) ----------------
__global__ __launch_bounds__(256)
void k_copy(const float* __restrict__ x, float* __restrict__ out) {
  size_t i = (size_t)blockIdx.x * 256 + threadIdx.x;
  ((F4*)out)[i] = ((const F4*)x)[i];
}

// ---------------- LDS staging helpers ----------------
__device__ __forceinline__ void stage_A_bf16(u16* sA, const u16* __restrict__ A,
                                             int m0, int kb, int K) {
  for (int i = threadIdx.x; i < 128 * 8; i += 256) {
    int r = i >> 3, c = (i & 7) * 8;
    *(Q4*)&sA[r * 64 + c] = *(const Q4*)&A[(size_t)(m0 + r) * K + kb + c];
  }
}
#if HAVE_ASYNC
__device__ __forceinline__ void stage_A_bf16_async(u16* sA, const u16* __restrict__ A,
                                                   int m0, int kb, int K) {
  for (int i = threadIdx.x; i < 128 * 8; i += 256) {
    int r = i >> 3, c = (i & 7) * 8;
    async_cp16(&A[(size_t)(m0 + r) * K + kb + c], &sA[r * 64 + c]);
  }
}
__device__ __forceinline__ void stage_BT_bf16_async(u16* sB, const u16* __restrict__ BT,
                                                    int n0, int kb, int K) {
  for (int i = threadIdx.x; i < 128 * 8; i += 256) {
    int n = i >> 3, c = (i & 7) * 8;
    async_cp16(&BT[(size_t)(n0 + n) * K + kb + c], &sB[n * 72 + c]);
  }
}
#endif
__device__ __forceinline__ void stage_A_f32(u16* sA, const float* __restrict__ A,
                                            int m0, int kb, int K) {
  for (int i = threadIdx.x; i < 128 * 16; i += 256) {
    int r = i >> 4, c = (i & 15) * 4;
    F4 f = *(const F4*)&A[(size_t)(m0 + r) * K + kb + c];
    *(u32*)&sA[r * 64 + c]     = (u32)f2bf(f.x) | ((u32)f2bf(f.y) << 16);
    *(u32*)&sA[r * 64 + c + 2] = (u32)f2bf(f.z) | ((u32)f2bf(f.w) << 16);
  }
}
// B tile transposed into LDS: sB[n*72 + k] = B[kb+k][n0+n], converting f32->bf16
__device__ __forceinline__ void stage_B_f32(u16* sB, const float* __restrict__ Bsrc,
                                            int kb, int n0, int N) {
  for (int i = threadIdx.x; i < 32 * 128; i += 256) {
    int n = i & 127, kk = i >> 7;
    float a0 = Bsrc[(size_t)(kb + 2 * kk)     * N + n0 + n];
    float a1 = Bsrc[(size_t)(kb + 2 * kk + 1) * N + n0 + n];
    *(u32*)&sB[n * 72 + 2 * kk] = (u32)f2bf(a0) | ((u32)f2bf(a1) << 16);
  }
}
// B provided already transposed (row n holds K values): straight copy
__device__ __forceinline__ void stage_BT_bf16(u16* sB, const u16* __restrict__ BT,
                                              int n0, int kb, int K) {
  for (int i = threadIdx.x; i < 128 * 8; i += 256) {
    int n = i >> 3, c = (i & 7) * 8;
    *(Q4*)&sB[n * 72 + c] = *(const Q4*)&BT[(size_t)(n0 + n) * K + kb + c];
  }
}

// one 128x128x64 MMA step: 8 waves, each wave a 64x32 sub-tile (8 accumulators)
__device__ __forceinline__ void mma_step(const u16* sA, const u16* sB, v8f acc[8],
                                         int waveM, int waveN, int lane) {
  const int khalf = lane >> 4;
  const int l16   = lane & 15;
#pragma unroll
  for (int kstep = 0; kstep < 2; ++kstep) {
    const int k0 = kstep * 32;
    v16bf a[4], bfr[2];
#pragma unroll
    for (int mt = 0; mt < 4; ++mt) {
      const u16* p = sA + (waveM * 64 + mt * 16 + l16) * 64 + k0 + khalf * 8;
      a[mt] = load_frag(p, p + 16);
    }
#pragma unroll
    for (int nt = 0; nt < 2; ++nt) {
      const u16* p = sB + (waveN * 32 + nt * 16 + l16) * 72 + k0 + khalf * 8;
      bfr[nt] = load_frag(p, p + 16);
    }
#pragma unroll
    for (int mt = 0; mt < 4; ++mt)
#pragma unroll
      for (int nt = 0; nt < 2; ++nt)
        acc[mt * 2 + nt] = __builtin_amdgcn_wmma_f32_16x16x32_bf16(
            false, a[mt], false, bfr[nt], (short)0, acc[mt * 2 + nt], false, false);
  }
}

// ---------------- generic single-B GEMM with fused epilogues ----------------
enum { EPI_STORE_BF16T = 0, EPI_ATTN = 1, EPI_DOWN = 2 };

template <int EPI, bool A_BF16, bool B_TBF16, int N_, int K_>
__global__ __launch_bounds__(256)
void gemm_sb(const void* __restrict__ Ap, const void* __restrict__ Bp,
             void* __restrict__ Cp, const int* __restrict__ cnt,
             const int* __restrict__ sel, const float* __restrict__ xadd,
             const float* __restrict__ resc, int Mmax) {
  const int n0 = blockIdx.x * 128;
  const int m0 = blockIdx.y * 128;
  const int mlimit = cnt ? min(*cnt, Mmax) : Mmax;
  if (m0 >= mlimit) return;

  __shared__ u16 sA[128 * 64];
  __shared__ u16 sB[128 * 72];

  const int lane  = threadIdx.x & 31;
  const int wid   = threadIdx.x >> 5;
  const int waveM = wid >> 2;
  const int waveN = wid & 3;

  v8f z = {0.f, 0.f, 0.f, 0.f, 0.f, 0.f, 0.f, 0.f};
  v8f acc[8];
#pragma unroll
  for (int i = 0; i < 8; ++i) acc[i] = z;

  for (int kb = 0; kb < K_; kb += 64) {
    if (kb + 64 < K_) {  // global_prefetch_b8 for next K tile
      if (A_BF16) __builtin_prefetch(&((const u16*)Ap)[(size_t)(m0 + (threadIdx.x >> 1)) * K_ + kb + 64], 0, 1);
      else        __builtin_prefetch(&((const float*)Ap)[(size_t)(m0 + (threadIdx.x >> 1)) * K_ + kb + 64], 0, 1);
      if (B_TBF16) __builtin_prefetch(&((const u16*)Bp)[(size_t)(n0 + (threadIdx.x >> 1)) * K_ + kb + 64], 0, 1);
      else         __builtin_prefetch(&((const float*)Bp)[(size_t)(kb + 64 + (threadIdx.x >> 3)) * N_ + n0], 0, 1);
    }
    if constexpr (A_BF16) {
#if HAVE_ASYNC
      stage_A_bf16_async(sA, (const u16*)Ap, m0, kb, K_);
#else
      stage_A_bf16(sA, (const u16*)Ap, m0, kb, K_);
#endif
    } else {
      stage_A_f32(sA, (const float*)Ap, m0, kb, K_);
    }
    if constexpr (B_TBF16) {
#if HAVE_TDM
      if (wid == 0) tdm_load_bt(sB, (const u16*)Bp, n0, kb, K_, N_);
#elif HAVE_ASYNC
      stage_BT_bf16_async(sB, (const u16*)Bp, n0, kb, K_);
#else
      stage_BT_bf16(sB, (const u16*)Bp, n0, kb, K_);
#endif
    } else {
      stage_B_f32(sB, (const float*)Bp, kb, n0, N_);
    }
#if HAVE_ASYNC
    if constexpr (A_BF16) wait_async0();
#endif
#if HAVE_TDM
    if constexpr (B_TBF16) {
      if (wid == 0) __builtin_amdgcn_s_wait_tensorcnt(0);
    }
#endif
    __syncthreads();
    mma_step(sA, sB, acc, waveM, waveN, lane);
    __syncthreads();
  }

#pragma unroll
  for (int mt = 0; mt < 4; ++mt) {
#pragma unroll
    for (int nt = 0; nt < 2; ++nt) {
      v8f c = acc[mt * 2 + nt];
      int gmBase = m0 + waveM * 64 + mt * 16 + (lane >> 4) * 8;
      int gn     = n0 + waveN * 32 + nt * 16 + (lane & 15);
#pragma unroll
      for (int r = 0; r < 8; ++r) {
        int gm = gmBase + r;
        if (gm >= mlimit) continue;
        if constexpr (EPI == EPI_STORE_BF16T) {
          // store TRANSPOSED so the consumer GEMM can raw-copy / TDM its B tiles
          ((u16*)Cp)[(size_t)gn * Mmax + gm] = f2bf(c[r]);
        } else if constexpr (EPI == EPI_ATTN) {
          int tok = sel[gm];
          ((float*)Cp)[(size_t)gm * N_ + gn] = c[r] + xadd[(size_t)tok * N_ + gn];
        } else {  // EPI_DOWN
          int tok = sel[gm];
          ((float*)Cp)[(size_t)tok * N_ + gn] = resc[(size_t)gm * N_ + gn] + c[r];
        }
      }
    }
  }
}

// ---------------- fused gate+up GEMM: h = silu(n2@Wg) * (n2@Wu) ----------------
__global__ __launch_bounds__(256)
void k_gateup(const u16* __restrict__ A, const float* __restrict__ Bg,
              const float* __restrict__ Bu, u16* __restrict__ hc,
              const int* __restrict__ cnt) {
  const int n0 = blockIdx.x * 128;
  const int m0 = blockIdx.y * 128;
  const int mlimit = min(*cnt, BS);
  if (m0 >= mlimit) return;

  __shared__ u16 sA[128 * 64];
  __shared__ u16 sBg[128 * 72];
  __shared__ u16 sBu[128 * 72];

  const int lane  = threadIdx.x & 31;
  const int wid   = threadIdx.x >> 5;
  const int waveM = wid >> 2;
  const int waveN = wid & 3;

  v8f z = {0.f, 0.f, 0.f, 0.f, 0.f, 0.f, 0.f, 0.f};
  v8f accg[8], accu[8];
#pragma unroll
  for (int i = 0; i < 8; ++i) { accg[i] = z; accu[i] = z; }

  for (int kb = 0; kb < H; kb += 64) {
    if (kb + 64 < H) {
      __builtin_prefetch(&Bg[(size_t)(kb + 64 + (threadIdx.x >> 3)) * DFF + n0], 0, 1);
      __builtin_prefetch(&Bu[(size_t)(kb + 64 + (threadIdx.x >> 3)) * DFF + n0], 0, 1);
    }
#if HAVE_ASYNC
    stage_A_bf16_async(sA, A, m0, kb, H);
#else
    stage_A_bf16(sA, A, m0, kb, H);
#endif
    stage_B_f32(sBg, Bg, kb, n0, DFF);
    stage_B_f32(sBu, Bu, kb, n0, DFF);
#if HAVE_ASYNC
    wait_async0();
#endif
    __syncthreads();
    mma_step(sA, sBg, accg, waveM, waveN, lane);
    mma_step(sA, sBu, accu, waveM, waveN, lane);
    __syncthreads();
  }

#pragma unroll
  for (int mt = 0; mt < 4; ++mt) {
#pragma unroll
    for (int nt = 0; nt < 2; ++nt) {
      v8f g = accg[mt * 2 + nt];
      v8f u = accu[mt * 2 + nt];
      int gmBase = m0 + waveM * 64 + mt * 16 + (lane >> 4) * 8;
      int gn     = n0 + waveN * 32 + nt * 16 + (lane & 15);
#pragma unroll
      for (int r = 0; r < 8; ++r) {
        int gm = gmBase + r;
        if (gm >= mlimit) continue;
        float gv = g[r];
        float s  = gv / (1.f + __expf(-gv));   // silu
        hc[(size_t)gm * DFF + gn] = f2bf(s * u[r]);
      }
    }
  }
}

// ---------------- launch ----------------
extern "C" void kernel_launch(void* const* d_in, const int* in_sizes, int n_in,
                              void* d_out, int out_size, void* d_ws, size_t ws_size,
                              hipStream_t stream) {
  const float* x   = (const float*)d_in[0];
  const float* wr  = (const float*)d_in[1];
  const float* br  = (const float*)d_in[2];
  // d_in[3]=Wq, d_in[4]=Wk are provably unused (softmax over singleton axis == 1)
  const float* Wv  = (const float*)d_in[5];
  const float* Wo  = (const float*)d_in[6];
  const float* Wg  = (const float*)d_in[7];
  const float* Wu  = (const float*)d_in[8];
  const float* Wd  = (const float*)d_in[9];
  const float* n1w = (const float*)d_in[10];
  const float* n2w = (const float*)d_in[11];
  float* out = (float*)d_out;
  (void)in_sizes; (void)n_in; (void)out_size; (void)ws_size;

  char* ws = (char*)d_ws;
  size_t off = 0;
  auto wsalloc = [&](size_t bytes) -> void* {
    void* p = ws + off;
    off = (off + bytes + 255) & ~(size_t)255;
    return p;
  };
  float* weights = (float*)wsalloc((size_t)BS * 4);
  float* thresh  = (float*)wsalloc(256);
  int*   cnt     = (int*)  wsalloc(256);
  int*   sel     = (int*)  wsalloc((size_t)BS * 4);
  u16*   WvoT    = (u16*)  wsalloc((size_t)H * H * 2);
  u16*   n1c     = (u16*)  wsalloc((size_t)BS * H * 2);
  float* resc    = (float*)wsalloc((size_t)BS * H * 4);
  u16*   n2c     = (u16*)  wsalloc((size_t)BS * H * 2);
  u16*   hc      = (u16*)  wsalloc((size_t)BS * DFF * 2);

  // 1. router weights + zero selection counter
  k_router<<<BS, 256, 0, stream>>>(x, wr, br, weights, cnt);
  // 2. exact k-th largest threshold per batch
  k_select<<<Bc, 256, 0, stream>>>(weights, thresh);
  // 3. compact selected tokens, fused RMSNorm1 -> bf16
  k_compact_norm<<<BS, 256, 0, stream>>>(x, weights, thresh, n1w, cnt, sel, n1c);
  // 4. WvoT = (Wv @ Wo)^T (bf16) — folds degenerate attention into one GEMM,
  //    stored transposed so the next GEMM's B tiles are raw-copyable (TDM/async)
  gemm_sb<EPI_STORE_BF16T, false, false, H, H>
      <<<dim3(H / 128, H / 128), 256, 0, stream>>>(Wv, Wo, WvoT, nullptr, nullptr,
                                                   nullptr, nullptr, H);
  // 5. background: out = x for every token
  k_copy<<<(BS * H / 4) / 256, 256, 0, stream>>>(x, out);
  // 6. resc = x[sel] + n1c @ Wvo  (compacted residual, f32; B staged via TDM)
  gemm_sb<EPI_ATTN, true, true, H, H>
      <<<dim3(H / 128, BS / 128), 256, 0, stream>>>(n1c, WvoT, resc, cnt, sel, x,
                                                    nullptr, BS);
  // 7. n2c = rmsnorm(resc) (bf16)
  k_norm2<<<BS, 256, 0, stream>>>(resc, n2w, cnt, n2c);
  // 8. hc = silu(n2c@Wg) * (n2c@Wu) (bf16, fused dual GEMM)
  k_gateup<<<dim3(DFF / 128, BS / 128), 256, 0, stream>>>(n2c, Wg, Wu, hc, cnt);
  // 9. out[sel] = resc + hc @ Wd  (scatter over the x background)
  gemm_sb<EPI_DOWN, true, false, H, DFF>
      <<<dim3(H / 128, BS / 128), 256, 0, stream>>>(hc, Wd, out, cnt, sel, nullptr,
                                                    resc, BS);
}
